// GraphTextLanguageModel_36051955483066
// MI455X (gfx1250) — compile-verified, matrix operand order checked by reading, not run on hardware
//
#include <hip/hip_runtime.h>
#include <hip/hip_bf16.h>
#include <math.h>

// Problem constants (match reference)
#define B_ 64
#define T_ 12
#define D_ 256
#define S_ 512
#define V_ 512

typedef __attribute__((ext_vector_type(16))) _Float16 v16h;
typedef __attribute__((ext_vector_type(8)))  _Float16 v8h;
typedef __attribute__((ext_vector_type(8)))  float    v8f;

__device__ __forceinline__ float gelu_exact(float x) {
  // torch nn.GELU default: 0.5*x*(1+erf(x/sqrt(2)))
  return 0.5f * x * (1.0f + erff(x * 0.70710678118654752440f));
}
__device__ __forceinline__ float sigmoidf_(float x) {
  return 1.0f / (1.0f + __expf(-x));
}
__device__ __forceinline__ int clampi(int v, int hi) {
  return v < 0 ? 0 : (v > hi ? hi : v);
}

// ---- WMMA 16x16x32 f16 fragment loader (ISA 7.12.2 layout, wave32) ----
// Works for both operands given f16 data laid out with the fragment's
// 16-wide index as the ROW and K contiguous:
//   A: activations row-major [M][K];  B: weights transposed [N][K].
// lane l: r = row0 + (l&15); element i covers
//   k = kbase + (i/8)*16 + (l/16)*8 + (i%8)
// -> elements 0..7 and 8..15 are two contiguous 16-byte chunks.
__device__ __forceinline__ v16h load_frag16(const _Float16* base, int row0,
                                            int ld, int kbase, int lane) {
  const int r  = row0 + (lane & 15);
  const int hb = (lane >> 4) << 3;
  const _Float16* p = base + r * ld + kbase + hb;
  v8h lo = *(const v8h*)(p);
  v8h hi = *(const v8h*)(p + 16);
  return __builtin_shufflevector(lo, hi, 0, 1, 2, 3, 4, 5, 6, 7,
                                         8, 9, 10, 11, 12, 13, 14, 15);
}

// ============================================================================
// Prep kernel A: f32 -> f16 copy (evidence rows, already K-contiguous)
// ============================================================================
__global__ __launch_bounds__(256)
void cvt_f16_kernel(const float* __restrict__ in, _Float16* __restrict__ out,
                    int n)
{
  const int i = blockIdx.x * 256 + threadIdx.x;
  if (i < n) out[i] = (_Float16)in[i];
}

// ============================================================================
// Prep kernel B: transpose-convert weights [K][N] f32 -> [N][K] f16
// grid = (K/16, N/16), block = 256 (16x16 tile via LDS).
// ============================================================================
__global__ __launch_bounds__(256)
void transpose_f16_kernel(const float* __restrict__ in,
                          _Float16* __restrict__ out, int K, int N)
{
  __shared__ float tilebuf[16][17];
  const int k0 = blockIdx.x * 16, n0 = blockIdx.y * 16;
  const int tx = threadIdx.x & 15, ty = threadIdx.x >> 4;
  tilebuf[ty][tx] = in[(k0 + ty) * N + (n0 + tx)];
  __syncthreads();
  out[(n0 + ty) * K + (k0 + tx)] = (_Float16)tilebuf[tx][ty];
}

// ============================================================================
// Kernel 1: gate MLPs.  mg/sg[b*T+t] = sigmoid(gelu(evd @ W1 + b1) @ W2 + b2)
// [768,256] @ [256,256] GEMM per gate via WMMA; scalar head reduced in LDS.
// grid = (48 row-tiles, 2 gates), block = 128 (4 waves).
// ============================================================================
__global__ __launch_bounds__(128)
void gate_mlp_kernel(const _Float16* __restrict__ evh,      // [768][256] f16
                     const _Float16* __restrict__ W1Tg,     // [256][256] f16 (N,K)
                     const float* __restrict__ bg1, const float* __restrict__ Wg2,
                     const float* __restrict__ bg2,
                     const _Float16* __restrict__ W1Ts,
                     const float* __restrict__ bs1, const float* __restrict__ Ws2,
                     const float* __restrict__ bs2,
                     float* __restrict__ mg_out, float* __restrict__ sg_out)
{
  const int tile = blockIdx.x;
  const int g    = blockIdx.y;
  const _Float16* W1T = g ? W1Ts : W1Tg;
  const float* b1 = g ? bs1 : bg1;
  const float* W2 = g ? Ws2 : Wg2;
  const float* b2 = g ? bs2 : bg2;
  float* out = g ? sg_out : mg_out;

  const int lane = threadIdx.x & 31;
  const int wave = threadIdx.x >> 5;
  const int row0 = tile * 16;

  __shared__ float red[16];
  if (threadIdx.x < 16) red[threadIdx.x] = 0.0f;
  __syncthreads();

  // Preload all 8 A fragments (K = 256) for this 16-row tile (2 b128 each).
  v16h a[8];
  #pragma unroll
  for (int kk = 0; kk < 8; ++kk)
    a[kk] = load_frag16(evh, row0, D_, kk * 32, lane);

  float part[8];
  #pragma unroll
  for (int j = 0; j < 8; ++j) part[j] = 0.0f;

  for (int q = 0; q < 4; ++q) {
    const int nt = wave * 4 + q;                 // 16 n-tiles over 4 waves
    __builtin_prefetch(&W1T[(nt * 16 + 16) * D_], 0, 1);  // next tile's B rows
    v8f acc = {};
    #pragma unroll
    for (int kk = 0; kk < 8; ++kk) {
      v16h bfr = load_frag16(W1T, nt * 16, D_, kk * 32, lane);
      acc = __builtin_amdgcn_wmma_f32_16x16x32_f16(false, a[kk], false, bfr,
                                                   (short)0, acc, false, false);
    }
    const int n   = nt * 16 + (lane & 15);
    const float w2n = W2[n];     // W2 is (D,1)
    const float b1n = b1[n];
    #pragma unroll
    for (int j = 0; j < 8; ++j)
      part[j] += gelu_exact(acc[j] + b1n) * w2n;
  }
  const int mbase = (lane >> 4) << 3;
  #pragma unroll
  for (int j = 0; j < 8; ++j)
    atomicAdd(&red[mbase + j], part[j]);         // ds_add_f32
  __syncthreads();
  if (threadIdx.x < 16)
    out[row0 + threadIdx.x] = sigmoidf_(red[threadIdx.x] + b2[0]);
}

// ============================================================================
// Kernel 2: sparse graph memory + walk readout.  One block per batch.
// walk support is {q} ∪ {tsym_e} (≤13).  acc_v/acc_s collapse to weighted
// sums of ≤13 symbol_emb rows and ≤12 value_emb rows using Σ_iter walk_iter.
// Writes graph_state [B, 2D] to workspace as f16 (feeds head WMMA A operand).
// ============================================================================
__global__ __launch_bounds__(256)
void graph_walk_kernel(const int* __restrict__ em,  const int* __restrict__ si,
                       const int* __restrict__ sv,  const int* __restrict__ tsi,
                       const int* __restrict__ tsv, const int* __restrict__ tvi,
                       const int* __restrict__ tvv, const int* __restrict__ qi,
                       const int* __restrict__ qv,
                       const float* __restrict__ mg, const float* __restrict__ sg,
                       const float* __restrict__ symbol_emb,
                       const float* __restrict__ value_emb,
                       _Float16* __restrict__ gs_h)
{
  const int b = blockIdx.x;

  __shared__ int   sup[13];
  __shared__ float coeff[13];      // Σ over 4 walk iterations, on support
  __shared__ int   mp_tval[12];
  __shared__ float mp_w[12];       // gated map weight * walk coefficient
  __shared__ int   cnt[2];         // {nsup, nmap}
  // thread-0 scratch (in LDS to avoid private-segment spills)
  __shared__ float Ms[13 * 13];
  __shared__ int   st_i[12], st_j[12], mp_src[12];
  __shared__ float st_w[12], cbuf[13], cnbuf[13], csbuf[13];

  if (threadIdx.x == 0) {
    int ns = 0, nstep = 0, nmap = 0;
    const int q = clampi(qi[b], S_ - 1);
    sup[ns++] = q;
    for (int t = 0; t < T_; ++t) {
      const int idx = b * T_ + t;
      const int mk   = em[idx];
      const int src  = clampi(si[idx],  S_ - 1);
      const int tsym = clampi(tsi[idx], S_ - 1);
      const int tval = clampi(tvi[idx], V_ - 1);
      const bool svb = sv[idx] > 0, tsvb = tsv[idx] > 0, tvvb = tvv[idx] > 0;
      if ((mk == 0 || mk == 1) && svb && tvvb) {   // PASS | MAP
        mp_src[nmap] = src; mp_tval[nmap] = tval; mp_w[nmap] = mg[idx]; ++nmap;
      }
      if (mk == 2 && svb && tsvb) {                // STEP
        st_i[nstep] = src; st_j[nstep] = tsym; st_w[nstep] = sg[idx]; ++nstep;
        int f = -1;
        for (int i2 = 0; i2 < ns; ++i2) if (sup[i2] == tsym) { f = i2; break; }
        if (f < 0) sup[ns++] = tsym;
      }
    }
    // Dense step transition restricted to the support: Ms[j][i]
    for (int i2 = 0; i2 < ns * ns; ++i2) Ms[i2] = 0.0f;
    for (int e = 0; e < nstep; ++e) {
      int ii = -1, jj = -1;
      for (int i2 = 0; i2 < ns; ++i2) {
        if (sup[i2] == st_i[e]) ii = i2;
        if (sup[i2] == st_j[e]) jj = i2;
      }
      if (ii >= 0 && jj >= 0) Ms[jj * ns + ii] += st_w[e];
    }
    // walk0 = one-hot(q)*query_valid; 3 graph steps; coeff = Σ walks
    for (int i2 = 0; i2 < ns; ++i2) { cbuf[i2] = 0.0f; csbuf[i2] = 0.0f; }
    cbuf[0] = (qv[b] > 0) ? 1.0f : 0.0f;
    csbuf[0] = cbuf[0];
    for (int it = 0; it < 3; ++it) {
      for (int j2 = 0; j2 < ns; ++j2) {
        float s = 0.0f;
        for (int i2 = 0; i2 < ns; ++i2) s += Ms[j2 * ns + i2] * cbuf[i2];
        cnbuf[j2] = s;
      }
      for (int j2 = 0; j2 < ns; ++j2) { cbuf[j2] = cnbuf[j2]; csbuf[j2] += cnbuf[j2]; }
    }
    for (int i2 = 0; i2 < ns; ++i2) coeff[i2] = csbuf[i2];
    // Fold walk coefficient into each map event's weight (acc_v columns)
    for (int e = 0; e < nmap; ++e) {
      int ii = -1;
      for (int i2 = 0; i2 < ns; ++i2) if (sup[i2] == mp_src[e]) { ii = i2; break; }
      mp_w[e] = (ii >= 0) ? csbuf[ii] * mp_w[e] : 0.0f;
    }
    cnt[0] = ns; cnt[1] = nmap;
  }
  __syncthreads();

  const int ns = cnt[0], nmap = cnt[1];
  const int d = threadIdx.x;                      // 0..255
  float s_acc = 0.0f, v_acc = 0.0f;
  for (int i = 0; i < ns; ++i)
    s_acc += coeff[i] * symbol_emb[sup[i] * D_ + d];          // acc_s
  for (int e = 0; e < nmap; ++e)
    v_acc += mp_w[e] * value_emb[mp_tval[e] * D_ + d];        // acc_v @ value_emb
  gs_h[b * (2 * D_) + d]      = (_Float16)s_acc;
  gs_h[b * (2 * D_) + D_ + d] = (_Float16)v_acc;
}

// ============================================================================
// Kernel 3: output heads.  logits = mlp2(gs, Wo*), feedback = mlp2(gs, Wf*).
// GEMM1 [16,512]@[512,256] -> gelu -> LDS(f16) -> GEMM2 [16,256]@[256,Nout].
// grid = (4 row-tiles, 2 heads), block = 128 (4 waves).
// ============================================================================
__global__ __launch_bounds__(128)
void head_mlp_kernel(const _Float16* __restrict__ gs_h,     // [64][512] f16
                     const _Float16* __restrict__ Wo1T,     // [256][512] f16
                     const float* __restrict__ bo1,
                     const _Float16* __restrict__ Wo2T,     // [512][256] f16
                     const float* __restrict__ bo2,
                     const _Float16* __restrict__ Wf1T,     // [256][512] f16
                     const float* __restrict__ bf1,
                     const _Float16* __restrict__ Wf2T,     // [256][256] f16
                     const float* __restrict__ bf2,
                     float* __restrict__ out_logits, float* __restrict__ out_feedback)
{
  const int tile = blockIdx.x;
  const int head = blockIdx.y;            // 0 = logits (V), 1 = feedback (D)
  const _Float16* W1T = head ? Wf1T : Wo1T;
  const float* b1 = head ? bf1 : bo1;
  const _Float16* W2T = head ? Wf2T : Wo2T;
  const float* b2 = head ? bf2 : bo2;
  float* out = head ? out_feedback : out_logits;
  const int Nout = head ? D_ : V_;

  const int lane = threadIdx.x & 31;
  const int wave = threadIdx.x >> 5;
  const int row0 = tile * 16;
  const int mbase = (lane >> 4) << 3;

  __shared__ __align__(16) _Float16 hid[16 * D_];   // gelu'd hidden tile, f16

  // GEMM1: K = 2D = 512, N = 256
  for (int nt = wave; nt < 16; nt += 4) {
    v8f acc = {};
    #pragma unroll
    for (int kk = 0; kk < 16; ++kk) {
      v16h afr = load_frag16(gs_h, row0, 2 * D_, kk * 32, lane);
      v16h bfr = load_frag16(W1T, nt * 16, 2 * D_, kk * 32, lane);
      acc = __builtin_amdgcn_wmma_f32_16x16x32_f16(false, afr, false, bfr,
                                                   (short)0, acc, false, false);
    }
    const int n = nt * 16 + (lane & 15);
    const float b1n = b1[n];
    #pragma unroll
    for (int j = 0; j < 8; ++j)
      hid[(mbase + j) * D_ + n] = (_Float16)gelu_exact(acc[j] + b1n);
  }
  __syncthreads();

  // GEMM2: K = 256, N = Nout (512 or 256); A streamed from LDS (ds_load_b128)
  for (int nt = wave; nt < (Nout >> 4); nt += 4) {
    v8f acc = {};
    #pragma unroll
    for (int kk = 0; kk < 8; ++kk) {
      v16h afr = load_frag16(hid, 0, D_, kk * 32, lane);
      v16h bfr = load_frag16(W2T, nt * 16, D_, kk * 32, lane);
      acc = __builtin_amdgcn_wmma_f32_16x16x32_f16(false, afr, false, bfr,
                                                   (short)0, acc, false, false);
    }
    const int col = nt * 16 + (lane & 15);
    const float b2c = b2[col];
    #pragma unroll
    for (int j = 0; j < 8; ++j)
      out[(row0 + mbase + j) * Nout + col] = acc[j] + b2c;
  }
}

// ============================================================================
extern "C" void kernel_launch(void* const* d_in, const int* in_sizes, int n_in,
                              void* d_out, int out_size, void* d_ws, size_t ws_size,
                              hipStream_t stream)
{
  const int*   em   = (const int*)d_in[0];
  const int*   si   = (const int*)d_in[1];
  const int*   sv   = (const int*)d_in[2];
  const int*   tsi  = (const int*)d_in[3];
  const int*   tsv  = (const int*)d_in[4];
  const int*   tvi  = (const int*)d_in[5];
  const int*   tvv  = (const int*)d_in[6];
  const int*   qi   = (const int*)d_in[7];
  const int*   qv   = (const int*)d_in[8];
  const float* evd  = (const float*)d_in[9];
  const float* semb = (const float*)d_in[10];
  const float* vemb = (const float*)d_in[11];
  const float* Wmg1 = (const float*)d_in[12]; const float* bmg1 = (const float*)d_in[13];
  const float* Wmg2 = (const float*)d_in[14]; const float* bmg2 = (const float*)d_in[15];
  const float* Wsg1 = (const float*)d_in[16]; const float* bsg1 = (const float*)d_in[17];
  const float* Wsg2 = (const float*)d_in[18]; const float* bsg2 = (const float*)d_in[19];
  const float* Wf1  = (const float*)d_in[20]; const float* bf1  = (const float*)d_in[21];
  const float* Wf2  = (const float*)d_in[22]; const float* bf2  = (const float*)d_in[23];
  const float* Wo1  = (const float*)d_in[24]; const float* bo1  = (const float*)d_in[25];
  const float* Wo2  = (const float*)d_in[26]; const float* bo2  = (const float*)d_in[27];

  // ---- workspace layout (byte offsets, 256B-aligned blocks) ----
  char* ws = (char*)d_ws;
  float*    mg   = (float*)(ws + 0);               //   768 f32
  float*    sg   = (float*)(ws + 3072);            //   768 f32
  _Float16* evh  = (_Float16*)(ws + 8192);         // [768][256] f16   (384 KB)
  _Float16* W1Tg = (_Float16*)(ws + 401408);       // [256][256] f16   (128 KB)
  _Float16* W1Ts = (_Float16*)(ws + 532480);       // [256][256] f16   (128 KB)
  _Float16* Wo1T = (_Float16*)(ws + 663552);       // [256][512] f16   (256 KB)
  _Float16* Wf1T = (_Float16*)(ws + 925696);       // [256][512] f16   (256 KB)
  _Float16* Wo2T = (_Float16*)(ws + 1187840);      // [512][256] f16   (256 KB)
  _Float16* Wf2T = (_Float16*)(ws + 1449984);      // [256][256] f16   (128 KB)
  _Float16* gs_h = (_Float16*)(ws + 1581056);      // [64][512]  f16   ( 64 KB)

  float* out_logits   = (float*)d_out;             // [64, 512]
  float* out_feedback = out_logits + B_ * V_;      // [64, 256]

  // ---- stage 0: pre-pack activations / weights into WMMA-friendly f16 ----
  cvt_f16_kernel<<<768, 256, 0, stream>>>(evd, evh, B_ * T_ * D_);
  transpose_f16_kernel<<<dim3(16, 16), 256, 0, stream>>>(Wmg1, W1Tg, D_, D_);
  transpose_f16_kernel<<<dim3(16, 16), 256, 0, stream>>>(Wsg1, W1Ts, D_, D_);
  transpose_f16_kernel<<<dim3(32, 16), 256, 0, stream>>>(Wo1, Wo1T, 2 * D_, D_);
  transpose_f16_kernel<<<dim3(32, 16), 256, 0, stream>>>(Wf1, Wf1T, 2 * D_, D_);
  transpose_f16_kernel<<<dim3(16, 32), 256, 0, stream>>>(Wo2, Wo2T, D_, V_);
  transpose_f16_kernel<<<dim3(16, 16), 256, 0, stream>>>(Wf2, Wf2T, D_, D_);

  // ---- stage 1: gate MLPs (WMMA) ----
  gate_mlp_kernel<<<dim3(48, 2), 128, 0, stream>>>(
      evh, W1Tg, bmg1, Wmg2, bmg2, W1Ts, bsg1, Wsg2, bsg2, mg, sg);

  // ---- stage 2: sparse graph walk -> graph_state (f16) ----
  graph_walk_kernel<<<B_, 256, 0, stream>>>(
      em, si, sv, tsi, tsv, tvi, tvv, qi, qv, mg, sg, semb, vemb, gs_h);

  // ---- stage 3: output heads (WMMA) ----
  head_mlp_kernel<<<dim3(4, 2), 128, 0, stream>>>(
      gs_h, Wo1T, bo1, Wo2T, bo2, Wf1T, bf1, Wf2T, bf2,
      out_logits, out_feedback);
}
